// GroupedQueryAttention_20796231647735
// MI455X (gfx1250) — matrix operand order repaired and checked by reading, not compiled, not run on hardware
//
#include <hip/hip_runtime.h>

// ---------------------------------------------------------------------------
// GQA attention for MI455X (gfx1250, wave32, WMMA).
// B=2, T=2048, DIM=1024, NH=16, NKV=4, HD=64, R=4, causal.
// Pipeline: cvt(x)->bf16, pack weights into WMMA B-fragment layout,
// WMMA GEMM Q/KV projections, cooperative flash-attention with WMMA QK^T/PV
// and double-buffered GLOBAL_LOAD_ASYNC_TO_LDS staging of K/V tiles,
// WMMA GEMM output projection (fp32 out).
// ---------------------------------------------------------------------------

#define Bc 2
#define Tc 2048
#define DIMc 1024
#define NHc 16
#define NKVc 4
#define HDc 64

typedef __attribute__((ext_vector_type(16))) __bf16 v16bf;
typedef __attribute__((ext_vector_type(8)))  __bf16 v8bf;
typedef __attribute__((ext_vector_type(8)))  float  v8f;

// Async global->LDS copy of 16 bytes (CDNA5, tracked by ASYNCcnt).
// ldsoff = wave-relative LDS byte address (low 32 bits of generic pointer).
#define ASYNC_CP16(ldsoff, gptr)                                              \
  asm volatile("global_load_async_to_lds_b128 %0, %1, off"                    \
               :: "v"(ldsoff), "v"(gptr) : "memory")
#define WAIT_ASYNC0() asm volatile("s_wait_asynccnt 0x0" ::: "memory")

__device__ __forceinline__ unsigned lds_off(const void* p) {
  return (unsigned)(size_t)p;  // LDS aperture: low 32 bits == LDS byte offset
}

// ---------------------------------------------------------------------------
// fp32 -> bf16 elementwise convert
// ---------------------------------------------------------------------------
__global__ __launch_bounds__(256) void cvt_bf16_kernel(const float* __restrict__ in,
                                                       __bf16* __restrict__ out, int n) {
  int i = blockIdx.x * 256 + threadIdx.x;
  if (i < n) out[i] = (__bf16)in[i];
}

// ---------------------------------------------------------------------------
// Pack W[K,N] (fp32 row-major) into WMMA bf16 B-fragment layout:
// Wp[nt][kt][lane][e], tile = 32(K) x 16(N).
//   lane = (n%16) + ((k%32)>=16 ? 16 : 0),  e = (k%32)%16
// ---------------------------------------------------------------------------
__global__ __launch_bounds__(256) void pack_w_kernel(const float* __restrict__ W,
                                                     __bf16* __restrict__ Wp,
                                                     int K, int N) {
  int i = blockIdx.x * 256 + threadIdx.x;
  if (i >= K * N) return;
  int k = i / N, n = i % N;
  int kt = k >> 5, nt = n >> 4;
  int kq = k & 31, nq = n & 15;
  int lane = nq + ((kq >= 16) ? 16 : 0);
  int e = kq & 15;
  int KT = K >> 5;
  Wp[((((size_t)nt * KT) + kt) * 32 + lane) * 16 + e] = (__bf16)W[i];
}

// ---------------------------------------------------------------------------
// WMMA GEMM: C[M,N] = A[M,K](bf16 row-major) * W(pre-packed B-frags)
// 256 threads = 8 waves; wave tile 16(M) x 64(N); block tile 64 x 128.
// MODE 0: store bf16 Q as [B,NH,T,HD]
// MODE 1: N=512; first 256 cols -> K [B,NKV,T,HD]; last 256 -> V^T [B,NKV,HD,T]
// MODE 2: store fp32 row-major [M,N] (final output)
// ---------------------------------------------------------------------------
template <int MODE>
__global__ __launch_bounds__(256) void gemm_wmma_kernel(const __bf16* __restrict__ A,
                                                        const __bf16* __restrict__ Wp,
                                                        void* __restrict__ outA,
                                                        void* __restrict__ outB,
                                                        int M, int N, int K) {
  const int lane = threadIdx.x & 31;
  const int wave = threadIdx.x >> 5;
  const int wrow = wave & 3;
  const int wcol = wave >> 2;
  const int m0 = blockIdx.x * 64 + wrow * 16;
  const int n0 = blockIdx.y * 128 + wcol * 64;
  const int KT = K >> 5;
  const int laneHi = lane >> 4;
  const int lm = lane & 15;

  v8f acc[4] = {};

  const __bf16* Arow = A + (size_t)(m0 + lm) * K + (laneHi ? 16 : 0);

  for (int kt = 0; kt < KT; ++kt) {
    v16bf a;
    const __bf16* ap = Arow + kt * 32;
    ((v8bf*)&a)[0] = *(const v8bf*)(ap);
    ((v8bf*)&a)[1] = *(const v8bf*)(ap + 8);
    __builtin_prefetch(ap + 128, 0, 0);  // next-4-kt A cacheline
#pragma unroll
    for (int j = 0; j < 4; ++j) {
      const __bf16* bp = Wp + ((((size_t)(n0 >> 4) + j) * KT + kt) * 32 + lane) * 16;
      v16bf bfrag;
      ((v8bf*)&bfrag)[0] = *(const v8bf*)(bp);
      ((v8bf*)&bfrag)[1] = *(const v8bf*)(bp + 8);
      acc[j] = __builtin_amdgcn_wmma_f32_16x16x32_bf16(
          false, a, false, bfrag, (short)0, acc[j], false, false);
    }
  }

#pragma unroll
  for (int j = 0; j < 4; ++j) {
#pragma unroll
    for (int r = 0; r < 8; ++r) {
      int m = m0 + r + laneHi * 8;
      int n = n0 + j * 16 + lm;
      float v = acc[j][r];
      if (MODE == 2) {
        ((float*)outA)[(size_t)m * N + n] = v;
      } else if (MODE == 0) {
        int b = m >> 11, t = m & (Tc - 1);
        int h = n >> 6, hd = n & 63;
        ((__bf16*)outA)[((((size_t)b * NHc) + h) * Tc + t) * HDc + hd] = (__bf16)v;
      } else {  // MODE 1: KV
        int b = m >> 11, t = m & (Tc - 1);
        if (n < NKVc * HDc) {
          int kv = n >> 6, hd = n & 63;
          ((__bf16*)outA)[((((size_t)b * NKVc) + kv) * Tc + t) * HDc + hd] = (__bf16)v;
        } else {
          int vn = n - NKVc * HDc;
          int kv = vn >> 6, hd = vn & 63;
          ((__bf16*)outB)[((((size_t)b * NKVc) + kv) * HDc + hd) * Tc + t] = (__bf16)v;
        }
      }
    }
  }
}

// ---------------------------------------------------------------------------
// Cooperative flash attention: grid (T/64, NH, B), 128 threads = 4 waves.
// Block stages each 32-key K tile (32x64) and V^T tile (64x32) into LDS once
// via async global->LDS (double buffered); all 4 waves share them.
// Each wave owns 16 query rows; online softmax in C-fragment layout.
// ---------------------------------------------------------------------------
__global__ __launch_bounds__(128) void fattn_kernel(const __bf16* __restrict__ Qb,
                                                    const __bf16* __restrict__ Kb,
                                                    const __bf16* __restrict__ Vtb,
                                                    __bf16* __restrict__ Attn) {
  __shared__ __bf16 kbuf[2][32][64];  // [buf][key][hd]      4KB each
  __shared__ __bf16 vbuf[2][64][32];  // [buf][hd][key]      4KB each
  __shared__ __bf16 plds[4][16][32];  // per-wave P scratch  4KB total

  const int tid = threadIdx.x;
  const int lane = tid & 31;
  const int wave = tid >> 5;
  const int laneHi = lane >> 4;
  const int lm = lane & 15;
  const int h = blockIdx.y;
  const int b = blockIdx.z;
  const int q0b = blockIdx.x * 64;
  const int q0 = q0b + wave * 16;
  const int g = h >> 2;  // kv head (R = 4)

  const __bf16* Q  = Qb  + ((((size_t)b * NHc)  + h) * Tc + q0) * HDc;
  const char* Kp8  = (const char*)(Kb  + (((size_t)b * NKVc) + g) * Tc * HDc);
  const char* Vt8  = (const char*)(Vtb + (((size_t)b * NKVc) + g) * HDc * Tc);

  // Q A-fragments for the two hd k-steps (0..31, 32..63)
  v16bf qa[2];
#pragma unroll
  for (int kk = 0; kk < 2; ++kk) {
    const __bf16* ap = Q + lm * HDc + kk * 32 + (laneHi ? 16 : 0);
    ((v8bf*)&qa[kk])[0] = *(const v8bf*)(ap);
    ((v8bf*)&qa[kk])[1] = *(const v8bf*)(ap + 8);
  }

  float mrow[8], lrow[8];
  v8f O[4] = {};
#pragma unroll
  for (int r = 0; r < 8; ++r) { mrow[r] = -1e30f; lrow[r] = 0.0f; }

  const float scale = 0.125f;  // 1/sqrt(64)
  const int ntiles = (q0b + 64) / 32;  // uniform across block (causal max)

  const unsigned kofs0 = lds_off(&kbuf[0][0][0]);
  const unsigned kofs1 = lds_off(&kbuf[1][0][0]);
  const unsigned vofs0 = lds_off(&vbuf[0][0][0]);
  const unsigned vofs1 = lds_off(&vbuf[1][0][0]);

  // Stage one 32-key tile: K tile is 4KB linear; V^T tile is 64 rows x 64B.
  // 128 threads x 2 chunks of 16B each for K and for V.
  auto stage_tile = [&](int s0, int bufsel) {
    unsigned kofs = bufsel ? kofs1 : kofs0;
    unsigned vofs = bufsel ? vofs1 : vofs0;
#pragma unroll
    for (int c = 0; c < 2; ++c) {
      int ci = tid * 2 + c;  // 0..255
      ASYNC_CP16(kofs + ci * 16, Kp8 + (size_t)s0 * 128 + ci * 16);
      int hd = ci >> 2, part = ci & 3;
      ASYNC_CP16(vofs + ci * 16,
                 Vt8 + ((size_t)hd * Tc + s0) * 2 + part * 16);
    }
  };

  stage_tile(0, 0);

  for (int i = 0; i < ntiles; ++i) {
    const int s0 = i * 32;
    const int bufsel = i & 1;

    WAIT_ASYNC0();       // this thread's tile-i chunks are in LDS
    __syncthreads();     // everyone's chunks are in LDS; prev buf free

    if (i + 1 < ntiles) stage_tile(s0 + 32, bufsel ^ 1);  // overlap w/ compute

    // ---- S = Q K^T for 2 x 16-key tiles (K from LDS) -------------------
    v8f S[2] = {};
#pragma unroll
    for (int st = 0; st < 2; ++st) {
#pragma unroll
      for (int kk = 0; kk < 2; ++kk) {
        const __bf16* bp = &kbuf[bufsel][st * 16 + lm][kk * 32 + (laneHi ? 16 : 0)];
        v16bf kf;
        ((v8bf*)&kf)[0] = *(const v8bf*)(bp);
        ((v8bf*)&kf)[1] = *(const v8bf*)(bp + 8);
        S[st] = __builtin_amdgcn_wmma_f32_16x16x32_bf16(
            false, qa[kk], false, kf, (short)0, S[st], false, false);
      }
    }

    // ---- scale + causal mask + online softmax (per C-layout row) -------
    // Fully masked tiles (rows past causal limit) are neutral: corr=1, p=0.
#pragma unroll
    for (int r = 0; r < 8; ++r) {
      int row = r + laneHi * 8;
      int qi = q0 + row;
      float s0v = S[0][r] * scale;
      float s1v = S[1][r] * scale;
      if (s0 + lm > qi)      s0v = -1e30f;
      if (s0 + 16 + lm > qi) s1v = -1e30f;

      float mx = fmaxf(s0v, s1v);
#pragma unroll
      for (int off = 1; off < 16; off <<= 1)
        mx = fmaxf(mx, __shfl_xor(mx, off, 32));
      float mnew = fmaxf(mrow[r], mx);
      float corr = __expf(mrow[r] - mnew);
      float p0 = __expf(s0v - mnew);
      float p1 = __expf(s1v - mnew);
      float rs = p0 + p1;
#pragma unroll
      for (int off = 1; off < 16; off <<= 1)
        rs += __shfl_xor(rs, off, 32);
      lrow[r] = lrow[r] * corr + rs;
      mrow[r] = mnew;
#pragma unroll
      for (int nt = 0; nt < 4; ++nt) O[nt][r] *= corr;

      plds[wave][row][lm]      = (__bf16)p0;
      plds[wave][row][lm + 16] = (__bf16)p1;
    }

    // ---- reload P as A-fragment (per-wave LDS) -------------------------
    v16bf pf;
    {
      const __bf16* lp = &plds[wave][lm][laneHi ? 16 : 0];
      ((v8bf*)&pf)[0] = *(const v8bf*)(lp);
      ((v8bf*)&pf)[1] = *(const v8bf*)(lp + 8);
    }

    // ---- O += P V (V^T tile in LDS -> contiguous B-frags) --------------
#pragma unroll
    for (int nt = 0; nt < 4; ++nt) {
      const __bf16* vp = &vbuf[bufsel][nt * 16 + lm][laneHi ? 16 : 0];
      v16bf vf;
      ((v8bf*)&vf)[0] = *(const v8bf*)(vp);
      ((v8bf*)&vf)[1] = *(const v8bf*)(vp + 8);
      O[nt] = __builtin_amdgcn_wmma_f32_16x16x32_bf16(
          false, pf, false, vf, (short)0, O[nt], false, false);
    }

    __syncthreads();  // all waves done reading bufsel before it is refilled
  }

  // ---- finalize: O /= l, store bf16 to [B,T,NH,HD] ---------------------
#pragma unroll
  for (int r = 0; r < 8; ++r) {
    float inv = 1.0f / lrow[r];
    int t = q0 + r + laneHi * 8;
    __bf16* op = Attn + ((((size_t)b * Tc) + t) * NHc + h) * HDc;
#pragma unroll
    for (int nt = 0; nt < 4; ++nt)
      op[nt * 16 + lm] = (__bf16)(O[nt][r] * inv);
  }
}

// ---------------------------------------------------------------------------
// Host launcher
// ---------------------------------------------------------------------------
extern "C" void kernel_launch(void* const* d_in, const int* in_sizes, int n_in,
                              void* d_out, int out_size, void* d_ws, size_t ws_size,
                              hipStream_t stream) {
  const float* x   = (const float*)d_in[0];
  const float* Wq  = (const float*)d_in[1];
  const float* Wkv = (const float*)d_in[2];
  const float* Wo  = (const float*)d_in[3];
  float* out = (float*)d_out;

  char* ws = (char*)d_ws;
  const size_t MB = 1ull << 20;
  __bf16* xb   = (__bf16*)(ws + 0);        //  8 MB  x bf16 [B*T, DIM]
  __bf16* Wqp  = (__bf16*)(ws + 8  * MB);  //  2 MB  packed Wq
  __bf16* Wkvp = (__bf16*)(ws + 10 * MB);  //  1 MB  packed Wkv
  __bf16* Wop  = (__bf16*)(ws + 11 * MB);  //  2 MB  packed Wo
  __bf16* Qb   = (__bf16*)(ws + 13 * MB);  //  8 MB  Q [B,NH,T,HD]
  __bf16* Kb   = (__bf16*)(ws + 21 * MB);  //  2 MB  K [B,NKV,T,HD]
  __bf16* Vtb  = (__bf16*)(ws + 23 * MB);  //  2 MB  V^T [B,NKV,HD,T]
  __bf16* Attn = (__bf16*)(ws + 25 * MB);  //  8 MB  attn out [B,T,NH,HD]

  const int M = Bc * Tc;  // 4096

  {
    int n = Bc * Tc * DIMc;
    cvt_bf16_kernel<<<n / 256, 256, 0, stream>>>(x, xb, n);
  }
  pack_w_kernel<<<(DIMc * DIMc) / 256, 256, 0, stream>>>(Wq, Wqp, DIMc, DIMc);
  pack_w_kernel<<<(DIMc * 2 * NKVc * HDc) / 256, 256, 0, stream>>>(Wkv, Wkvp, DIMc, 2 * NKVc * HDc);
  pack_w_kernel<<<(DIMc * DIMc) / 256, 256, 0, stream>>>(Wo, Wop, DIMc, DIMc);

  gemm_wmma_kernel<0><<<dim3(M / 64, DIMc / 128), 256, 0, stream>>>(
      xb, Wqp, (void*)Qb, nullptr, M, DIMc, DIMc);
  gemm_wmma_kernel<1><<<dim3(M / 64, (2 * NKVc * HDc) / 128), 256, 0, stream>>>(
      xb, Wkvp, (void*)Kb, (void*)Vtb, M, 2 * NKVc * HDc, DIMc);

  fattn_kernel<<<dim3(Tc / 64, NHc, Bc), 128, 0, stream>>>(Qb, Kb, Vtb, Attn);

  gemm_wmma_kernel<2><<<dim3(M / 64, DIMc / 128), 256, 0, stream>>>(
      Attn, Wop, (void*)out, nullptr, M, DIMc, DIMc);
}
